// VQ_77438260347296
// MI455X (gfx1250) — matrix operand order repaired and checked by reading, not compiled, not run on hardware
//
#include <hip/hip_runtime.h>
#include <hip/hip_bf16.h>

#define BB      32768
#define IN_DIM  512
#define HID     64
#define COMM    128
#define PDIM    2048
#define BETA    0.25f
#define KL_W    1.0f
#define ENT_W   0.1f
#define EPSI    1e-8f

typedef __attribute__((ext_vector_type(16))) _Float16 v16h;
typedef __attribute__((ext_vector_type(8)))  _Float16 v8h;
typedef __attribute__((ext_vector_type(8)))  float    v8f;
typedef __attribute__((address_space(3)))    _Float16 lds_f16;

__device__ __forceinline__ v8f wmma_f16(v16h a, v16h b, v8f c) {
    // D = A(16x32 f16) * B(32x16 f16) + C(16x16 f32)
    return __builtin_amdgcn_wmma_f32_16x16x32_f16(false, a, false, b, (short)0, c, false, false);
}

// A fragment (16x32, f16, row-major source, ld in halves). p points at row 0.
// ISA layout: lane L holds row L%16; halves[0:8]=K k0+(L>=16?8:0).. ; halves[8:16]=+16.
__device__ __forceinline__ v16h load_a_frag(const _Float16* p, int ld, int k0, int lane) {
    int r = lane & 15;
    int k = k0 + ((lane & 16) ? 8 : 0);
    const _Float16* base = p + (size_t)r * ld + k;
    v8h lo = *(const v8h*)(base);
    v8h hi = *(const v8h*)(base + 16);
    v16h out;
#pragma unroll
    for (int i = 0; i < 8; ++i) { out[i] = lo[i]; out[i + 8] = hi[i]; }
    return out;
}

// Same but converting from f32 source on the fly.
__device__ __forceinline__ v16h load_a_frag_f32(const float* p, int ld, int k0, int lane) {
    int r = lane & 15;
    int k = k0 + ((lane & 16) ? 8 : 0);
    const float* base = p + (size_t)r * ld + k;
    v16h out;
#pragma unroll
    for (int i = 0; i < 8; ++i) { out[i] = (_Float16)base[i]; out[i + 8] = (_Float16)base[i + 16]; }
    return out;
}

// B fragment (32x16, f16) from a transposed [N][K] array (row n == column n of B).
// ISA layout: lane L holds col N=L%16; halves[0:8]=K k0+(L>=16?8:0).. ; halves[8:16]=+16.
__device__ __forceinline__ v16h load_b_frag(const _Float16* pT, int ld, int n0, int k0, int lane) {
    int n = n0 + (lane & 15);
    int k = k0 + ((lane & 16) ? 8 : 0);
    const _Float16* base = pT + (size_t)n * ld + k;
    v8h lo = *(const v8h*)(base);
    v8h hi = *(const v8h*)(base + 16);
    v16h out;
#pragma unroll
    for (int i = 0; i < 8; ++i) { out[i] = lo[i]; out[i + 8] = hi[i]; }
    return out;
}

// ---- CDNA5 async global->LDS copy helpers --------------------------------
__device__ __forceinline__ unsigned lds_off(const _Float16* p) {
    return (unsigned)(size_t)(lds_f16*)p;   // generic -> AS(3) addrspacecast gives LDS byte offset
}

// Copy 16KB (64 protos x 128 halves) chunk: 256 threads x 4 async b128 ops,
// each op moves 16B/lane directly into LDS, tracked by ASYNCcnt.
__device__ __forceinline__ void async_copy_16k(const _Float16* gsrc, unsigned ldst, int tid) {
#pragma unroll
    for (int j = 0; j < 4; ++j) {
        unsigned loff = ldst + (unsigned)(tid * 16 + j * 4096);
        const char* g = (const char*)gsrc + tid * 16 + j * 4096;
        asm volatile("global_load_async_to_lds_b128 %0, %1, off"
                     :: "v"(loff), "v"(g) : "memory");
    }
}

__device__ __forceinline__ void wait_async0() {
#if __has_builtin(__builtin_amdgcn_s_wait_asynccnt)
    __builtin_amdgcn_s_wait_asynccnt(0);
#else
    asm volatile("s_wait_asynccnt 0" ::: "memory");
#endif
}

// ---------------------------------------------------------------- prep kernel
__global__ __launch_bounds__(256) void k_prep(
    const float* W_emb, const float* W1, const float* W2,
    const float* W_mu, const float* W_var, const float* protos,
    _Float16* WembT, _Float16* W1T, _Float16* W2T,
    _Float16* WmuT, _Float16* WvarT, _Float16* protosh,
    float* p2, float* approx_probs, float* accs)
{
    int stride = gridDim.x * blockDim.x;
    int tid = blockIdx.x * blockDim.x + threadIdx.x;
    for (int i = tid; i < IN_DIM * HID; i += stride) {
        int k = i / HID, n = i % HID;
        WembT[n * IN_DIM + k] = (_Float16)W_emb[i];
    }
    for (int i = tid; i < HID * HID; i += stride) {
        int k = i / HID, n = i % HID;
        W1T[n * HID + k] = (_Float16)W1[i];
    }
    for (int i = tid; i < HID * COMM; i += stride) {
        int k = i / COMM, n = i % COMM;
        W2T[n * HID + k] = (_Float16)W2[i];
    }
    for (int i = tid; i < COMM * COMM; i += stride) {
        int k = i / COMM, n = i % COMM;
        WmuT[n * COMM + k]  = (_Float16)W_mu[i];
        WvarT[n * COMM + k] = (_Float16)W_var[i];
    }
    for (int i = tid; i < PDIM * COMM; i += stride) protosh[i] = (_Float16)protos[i];
    for (int r = tid; r < PDIM; r += stride) {
        float s = 0.f;
        for (int c = 0; c < COMM; ++c) { float v = protos[r * COMM + c]; s += v * v; }
        p2[r] = s;
    }
    for (int i = tid; i < PDIM; i += stride) approx_probs[i] = 0.f;
    if (tid < 8) accs[tid] = 0.f;
}

// ---------------------------------------------------------------- MLP kernel
// 256 threads = 8 waves; wave handles 16 rows. All matmuls via WMMA f16.
__global__ __launch_bounds__(256) void k_mlp(
    const float* x, const float* eps,
    const _Float16* WembT, const float* b_emb,
    const _Float16* W1T,  const float* b1,
    const _Float16* W2T,  const float* b2,
    const _Float16* WmuT, const float* b_mu,
    const _Float16* WvarT, const float* b_var,
    _Float16* sampleh, _Float16* muh,
    float* s2, float* m2, float* kld_acc)
{
    __shared__ __align__(32) _Float16 sh_h[8][16][HID];    // 16 KB  (h, then h1)
    __shared__ __align__(32) _Float16 sh_r[8][16][COMM];   // 32 KB  (r)
    __shared__ __align__(32) _Float16 sh_ts[8][16][16];    // 4 KB   sample tile
    __shared__ __align__(32) _Float16 sh_tm[8][16][16];    // 4 KB   mu tile

    const int wave = threadIdx.x >> 5;
    const int lane = threadIdx.x & 31;
    const int n15  = lane & 15;
    const int hi8  = (lane & 16) ? 8 : 0;
    const size_t rowbase = (size_t)blockIdx.x * 128 + wave * 16;

    // ---- stage 1: h = x @ W_emb + b_emb  (K=512, N=64)
    {
        v8f acc[4] = {};
        for (int kt = 0; kt < 16; ++kt) {
            v16h a = load_a_frag_f32(x + rowbase * IN_DIM, IN_DIM, kt * 32, lane);
#pragma unroll
            for (int nt = 0; nt < 4; ++nt) {
                v16h b = load_b_frag(WembT, IN_DIM, nt * 16, kt * 32, lane);
                acc[nt] = wmma_f16(a, b, acc[nt]);
            }
        }
#pragma unroll
        for (int nt = 0; nt < 4; ++nt) {
            float bias = b_emb[nt * 16 + n15];
#pragma unroll
            for (int i = 0; i < 8; ++i)
                sh_h[wave][i + hi8][nt * 16 + n15] = (_Float16)(acc[nt][i] + bias);
        }
    }
    __syncthreads();

    // ---- stage 2: h1 = relu(h @ W1 + b1)  (K=64, N=64), overwrite sh_h
    {
        v8f acc[4] = {};
#pragma unroll
        for (int kt = 0; kt < 2; ++kt) {
            v16h a = load_a_frag(&sh_h[wave][0][0], HID, kt * 32, lane);
#pragma unroll
            for (int nt = 0; nt < 4; ++nt) {
                v16h b = load_b_frag(W1T, HID, nt * 16, kt * 32, lane);
                acc[nt] = wmma_f16(a, b, acc[nt]);
            }
        }
        __syncthreads();
#pragma unroll
        for (int nt = 0; nt < 4; ++nt) {
            float bias = b1[nt * 16 + n15];
#pragma unroll
            for (int i = 0; i < 8; ++i)
                sh_h[wave][i + hi8][nt * 16 + n15] = (_Float16)fmaxf(acc[nt][i] + bias, 0.f);
        }
    }
    __syncthreads();

    // ---- stage 3: r = relu(h1 @ W2 + b2)  (K=64, N=128)
    for (int nt = 0; nt < 8; ++nt) {
        v8f acc = {};
#pragma unroll
        for (int kt = 0; kt < 2; ++kt) {
            v16h a = load_a_frag(&sh_h[wave][0][0], HID, kt * 32, lane);
            v16h b = load_b_frag(W2T, HID, nt * 16, kt * 32, lane);
            acc = wmma_f16(a, b, acc);
        }
        float bias = b2[nt * 16 + n15];
#pragma unroll
        for (int i = 0; i < 8; ++i)
            sh_r[wave][i + hi8][nt * 16 + n15] = (_Float16)fmaxf(acc[i] + bias, 0.f);
    }
    __syncthreads();

    // ---- stage 4: mu / logvar / sample, per-row stats  (K=128, N=128)
    float s2v[8] = {}, m2v[8] = {};
    float kld_part = 0.f;
    for (int nt = 0; nt < 8; ++nt) {
        v8f am = {}, av = {};
#pragma unroll
        for (int kt = 0; kt < 4; ++kt) {
            v16h a  = load_a_frag(&sh_r[wave][0][0], COMM, kt * 32, lane);
            v16h bm = load_b_frag(WmuT,  COMM, nt * 16, kt * 32, lane);
            am = wmma_f16(a, bm, am);
            v16h bv = load_b_frag(WvarT, COMM, nt * 16, kt * 32, lane);
            av = wmma_f16(a, bv, av);
        }
        int n = nt * 16 + n15;
        float bmu = b_mu[n], bvar = b_var[n];
#pragma unroll
        for (int i = 0; i < 8; ++i) {
            size_t row = rowbase + i + hi8;
            float mu = am[i] + bmu;
            float lv = av[i] + bvar;
            float ev = __expf(lv);
            float smp = mu + __expf(0.5f * lv) * eps[row * COMM + n];
            kld_part += -0.5f * (1.f + lv - mu * mu - ev);
            s2v[i] += smp * smp;
            m2v[i] += mu * mu;
            sh_ts[wave][i + hi8][n15] = (_Float16)smp;
            sh_tm[wave][i + hi8][n15] = (_Float16)mu;
        }
        __syncthreads();
        // coalesced 16B stores of the 16x16 tile
        {
            int r2 = lane >> 1, hb = (lane & 1) * 8;
            v8h ts = *(const v8h*)&sh_ts[wave][r2][hb];
            v8h tm = *(const v8h*)&sh_tm[wave][r2][hb];
            *(v8h*)&sampleh[(rowbase + r2) * COMM + nt * 16 + hb] = ts;
            *(v8h*)&muh[(rowbase + r2) * COMM + nt * 16 + hb]     = tm;
        }
        __syncthreads();
    }

    // reduce row stats across the 16 lanes of each half-group
#pragma unroll
    for (int d = 1; d < 16; d <<= 1) {
#pragma unroll
        for (int i = 0; i < 8; ++i) {
            s2v[i] += __shfl_xor(s2v[i], d, 32);
            m2v[i] += __shfl_xor(m2v[i], d, 32);
        }
    }
    if (n15 == 0) {
#pragma unroll
        for (int i = 0; i < 8; ++i) {
            size_t row = rowbase + i + hi8;
            s2[row] = s2v[i];
            m2[row] = m2v[i];
        }
    }
    // full-wave KLD reduction, one atomic per wave
#pragma unroll
    for (int d = 1; d <= 16; d <<= 1) kld_part += __shfl_xor(kld_part, d, 32);
    if (lane == 0) atomicAdd(kld_acc, kld_part);
}

// ---------------------------------------------------------------- distance kernel
// Codebook chunks (64 protos = 16KB) are DMA'd into double-buffered LDS with
// GLOBAL_LOAD_ASYNC_TO_LDS_B128 (ASYNCcnt), overlapping WMMA on the live buffer.
// Pass 1: argmin over protos of ||sample-p||^2 + entropy row-sums.
// Pass 2: recompute mu.proto, accumulate normalized column sums (approx_probs).
__global__ __launch_bounds__(256) void k_dist(
    const _Float16* sampleh, const _Float16* muh, const _Float16* protosh,
    const float* p2, const float* s2, const float* m2,
    int* idxv, float* approx_probs)
{
    __shared__ float s_col[PDIM];                          // 8 KB
    __shared__ __align__(64) _Float16 s_proto[2][64][COMM]; // 2 x 16 KB
    const int tid  = threadIdx.x;
    const int wave = tid >> 5;
    const int lane = tid & 31;
    const int n15  = lane & 15;
    const int hi8  = (lane & 16) ? 8 : 0;
    const size_t rowbase = (size_t)blockIdx.x * 128 + wave * 16;
    const int NCHUNK = PDIM / 64;

    for (int j = tid; j < PDIM; j += 256) s_col[j] = 0.f;

    unsigned lb0 = lds_off(&s_proto[0][0][0]);
    unsigned lb1 = lds_off(&s_proto[1][0][0]);

    // hold the wave's 16 rows of sample/mu as A fragments in registers
    v16h a_s[4], a_m[4];
#pragma unroll
    for (int kt = 0; kt < 4; ++kt) {
        a_s[kt] = load_a_frag(sampleh + rowbase * COMM, COMM, kt * 32, lane);
        a_m[kt] = load_a_frag(muh     + rowbase * COMM, COMM, kt * 32, lane);
    }
    float s2v[8], m2v[8], rs[8], best[8];
    int bidx[8];
#pragma unroll
    for (int i = 0; i < 8; ++i) {
        s2v[i] = s2[rowbase + i + hi8];
        m2v[i] = m2[rowbase + i + hi8];
        rs[i] = 0.f; best[i] = 3.4e38f; bidx[i] = 0;
    }

    // ---------------- pass 1: argmin + entropy row-sums
    __syncthreads();
    async_copy_16k(protosh, lb0, tid);
    for (int c = 0; c < NCHUNK; ++c) {
        wait_async0();
        __syncthreads();                       // chunk c resident in buf[c&1]
        if (c + 1 < NCHUNK)
            async_copy_16k(protosh + (size_t)(c + 1) * 64 * COMM,
                           (c + 1) & 1 ? lb1 : lb0, tid);
        const _Float16* bufp = &s_proto[c & 1][0][0];
#pragma unroll
        for (int t = 0; t < 4; ++t) {
            v8f accS = {}, accM = {};
#pragma unroll
            for (int kt = 0; kt < 4; ++kt) {
                v16h b = load_b_frag(bufp, COMM, t * 16, kt * 32, lane);
                accS = wmma_f16(a_s[kt], b, accS);
                accM = wmma_f16(a_m[kt], b, accM);
            }
            int n = c * 64 + t * 16 + n15;
            float p2n = p2[n];
#pragma unroll
            for (int i = 0; i < 8; ++i) {
                float dS = s2v[i] + p2n - 2.f * accS[i];
                if (dS < best[i]) { best[i] = dS; bidx[i] = n; }
                float dM = m2v[i] + p2n - 2.f * accM[i];
                rs[i] += __expf(-0.125f * dM) + EPSI;
            }
        }
        __syncthreads();                       // all waves done reading buf[c&1]
    }
    // reduce argmin + rowsum across half-group lanes
#pragma unroll
    for (int d = 1; d < 16; d <<= 1) {
#pragma unroll
        for (int i = 0; i < 8; ++i) {
            float ov = __shfl_xor(best[i], d, 32);
            int   oi = __shfl_xor(bidx[i], d, 32);
            if (ov < best[i] || (ov == best[i] && oi < bidx[i])) { best[i] = ov; bidx[i] = oi; }
            rs[i] += __shfl_xor(rs[i], d, 32);
        }
    }
    if (n15 == 0) {
#pragma unroll
        for (int i = 0; i < 8; ++i) idxv[rowbase + i + hi8] = bidx[i];
    }
    float rinv[8];
#pragma unroll
    for (int i = 0; i < 8; ++i) rinv[i] = 1.f / rs[i];   // butterfly left total in all lanes

    // ---------------- pass 2: normalized column sums
    async_copy_16k(protosh, lb0, tid);
    for (int c = 0; c < NCHUNK; ++c) {
        wait_async0();
        __syncthreads();
        if (c + 1 < NCHUNK)
            async_copy_16k(protosh + (size_t)(c + 1) * 64 * COMM,
                           (c + 1) & 1 ? lb1 : lb0, tid);
        const _Float16* bufp = &s_proto[c & 1][0][0];
#pragma unroll
        for (int t = 0; t < 4; ++t) {
            v8f accM = {};
#pragma unroll
            for (int kt = 0; kt < 4; ++kt) {
                v16h b = load_b_frag(bufp, COMM, t * 16, kt * 32, lane);
                accM = wmma_f16(a_m[kt], b, accM);
            }
            int n = c * 64 + t * 16 + n15;
            float p2n = p2[n];
            float csum = 0.f;
#pragma unroll
            for (int i = 0; i < 8; ++i) {
                float dM = m2v[i] + p2n - 2.f * accM[i];
                csum += (__expf(-0.125f * dM) + EPSI) * rinv[i];
            }
            csum += __shfl_xor(csum, 16, 32);
            if (lane < 16) atomicAdd(&s_col[n], csum);   // ds_add_f32
        }
        __syncthreads();
    }
    for (int j = tid; j < PDIM; j += 256) atomicAdd(&approx_probs[j], s_col[j]);
}

// ---------------------------------------------------------------- gather + MSE
__global__ __launch_bounds__(256) void k_gather(
    const float* protos, const int* idxv, const _Float16* muh,
    float* out, float* acc_commit)
{
    __shared__ float red[256];
    size_t i = (size_t)blockIdx.x * 256 + threadIdx.x;   // grid sized exactly B*COMM/256
    size_t row = i >> 7;
    int c = (int)(i & 127);
    float q = protos[(size_t)idxv[row] * COMM + c];
    out[i] = q;
    float mu = (float)muh[i];
    float d = (q - mu) * (q - mu);
    red[threadIdx.x] = d;
    __syncthreads();
    for (int s = 128; s > 0; s >>= 1) {
        if (threadIdx.x < s) red[threadIdx.x] += red[threadIdx.x + s];
        __syncthreads();
    }
    if (threadIdx.x == 0) atomicAdd(acc_commit, red[0]);
}

// ---------------------------------------------------------------- final scalars
__global__ __launch_bounds__(256) void k_final(
    const float* approx_probs, const float* accs, float* out_scalars)
{
    __shared__ float red[256];
    float e = 0.f;
    for (int j = threadIdx.x; j < PDIM; j += 256) {
        float p = approx_probs[j] * (1.f / (float)BB);
        e += -p * __logf(p);
    }
    red[threadIdx.x] = e;
    __syncthreads();
    for (int s = 128; s > 0; s >>= 1) {
        if (threadIdx.x < s) red[threadIdx.x] += red[threadIdx.x + s];
        __syncthreads();
    }
    if (threadIdx.x == 0) {
        float ent = red[0];
        float kld = accs[0] / (float)BB;
        float mse = accs[1] / ((float)BB * (float)COMM);
        float vq  = (1.f + BETA) * mse + ENT_W * ent;
        out_scalars[0] = KL_W * kld + vq;   // total_loss
        out_scalars[1] = kld;               // kld
    }
}

// ---------------------------------------------------------------- host launch
extern "C" void kernel_launch(void* const* d_in, const int* in_sizes, int n_in,
                              void* d_out, int out_size, void* d_ws, size_t ws_size,
                              hipStream_t stream) {
    const float* x      = (const float*)d_in[0];
    const float* eps    = (const float*)d_in[1];
    const float* W_emb  = (const float*)d_in[2];
    const float* b_emb  = (const float*)d_in[3];
    const float* W1     = (const float*)d_in[4];
    const float* b1     = (const float*)d_in[5];
    const float* W2     = (const float*)d_in[6];
    const float* b2     = (const float*)d_in[7];
    const float* W_mu   = (const float*)d_in[8];
    const float* b_mu   = (const float*)d_in[9];
    const float* W_var  = (const float*)d_in[10];
    const float* b_var  = (const float*)d_in[11];
    const float* protos = (const float*)d_in[12];
    float* out = (float*)d_out;

    char* ws = (char*)d_ws;
    size_t off = 0;
    auto take = [&](size_t bytes) {
        char* p = ws + off;
        off = (off + bytes + 255) & ~(size_t)255;
        return p;
    };
    _Float16* sampleh = (_Float16*)take((size_t)BB * COMM * 2);
    _Float16* muh     = (_Float16*)take((size_t)BB * COMM * 2);
    _Float16* protosh = (_Float16*)take((size_t)PDIM * COMM * 2);
    _Float16* WembT   = (_Float16*)take((size_t)HID * IN_DIM * 2);
    _Float16* W1T     = (_Float16*)take((size_t)HID * HID * 2);
    _Float16* W2T     = (_Float16*)take((size_t)COMM * HID * 2);
    _Float16* WmuT    = (_Float16*)take((size_t)COMM * COMM * 2);
    _Float16* WvarT   = (_Float16*)take((size_t)COMM * COMM * 2);
    float* p2   = (float*)take((size_t)PDIM * 4);
    float* s2   = (float*)take((size_t)BB * 4);
    float* m2   = (float*)take((size_t)BB * 4);
    int*   idxv = (int*)take((size_t)BB * 4);
    float* approx_probs = (float*)take((size_t)PDIM * 4);
    float* accs = (float*)take(256);   // [0]=kld sum, [1]=commit sum

    k_prep<<<1024, 256, 0, stream>>>(W_emb, W1, W2, W_mu, W_var, protos,
                                     WembT, W1T, W2T, WmuT, WvarT, protosh,
                                     p2, approx_probs, accs);
    k_mlp<<<BB / 128, 256, 0, stream>>>(x, eps, WembT, b_emb, W1T, b1, W2T, b2,
                                        WmuT, b_mu, WvarT, b_var,
                                        sampleh, muh, s2, m2, &accs[0]);
    k_dist<<<BB / 128, 256, 0, stream>>>(sampleh, muh, protosh, p2, s2, m2,
                                         idxv, approx_probs);
    k_gather<<<(BB * COMM) / 256, 256, 0, stream>>>(protos, idxv, muh, out, &accs[1]);
    k_final<<<1, 256, 0, stream>>>(approx_probs, accs, out + (size_t)BB * COMM);
}
